// QuantIndexer_60601988546894
// MI455X (gfx1250) — matrix operand order repaired and checked by reading, not compiled, not run on hardware
//
#include <hip/hip_runtime.h>
#include <hip/hip_bf16.h>

typedef __attribute__((ext_vector_type(16))) _Float16 v16h;
typedef __attribute__((ext_vector_type(8)))  _Float16 v8h;
typedef __attribute__((ext_vector_type(8)))  float    v8f;

#define DIM 4096
#define NH 32
#define HD 128
#define RD 64
#define QR_ 1536
#define S_ 2048
#define NEG_INF_MASK (-1.0e9f)
#define INV_SQRT_HD 0.08838834764831845f   // 1/sqrt(128)

#define WMMA_F16(a, b, c) \
    __builtin_amdgcn_wmma_f32_16x16x32_f16(false, (a), false, (b), (short)0, (c), false, false)

// ---------------------------------------------------------------------------
// 1) 4-bit symmetric weight fake-quant of wq_b_w (4096 x 1536) -> fp16
// ---------------------------------------------------------------------------
__global__ void qi_wq_quant(const float* __restrict__ wq, _Float16* __restrict__ wq16) {
    const int row = blockIdx.x;               // 0..4095
    __shared__ float red[256];
    float m = 0.0f;
    for (int i = threadIdx.x; i < QR_; i += 256)
        m = fmaxf(m, fabsf(wq[row * QR_ + i]));
    red[threadIdx.x] = m;
    __syncthreads();
    for (int off = 128; off > 0; off >>= 1) {
        if (threadIdx.x < off) red[threadIdx.x] = fmaxf(red[threadIdx.x], red[threadIdx.x + off]);
        __syncthreads();
    }
    float s = red[0] / 7.0f;
    if (s == 0.0f) s = 1.0f;
    const float inv_s = 1.0f / s;
    for (int i = threadIdx.x; i < QR_; i += 256) {
        float r = fminf(fmaxf(rintf(wq[row * QR_ + i] * inv_s), -7.0f), 7.0f);
        wq16[row * QR_ + i] = (_Float16)(r * s);
    }
}

// ---------------------------------------------------------------------------
// 1b) wk fp32 -> fp16 (128 x 4096), so kgemm B-fragments load directly
// ---------------------------------------------------------------------------
__global__ void qi_wk16(const float* __restrict__ wk, _Float16* __restrict__ wk16) {
    const int i = blockIdx.x * 256 + threadIdx.x;
    wk16[i] = (_Float16)wk[i];
}

// ---------------------------------------------------------------------------
// helper: stage a 16 x 128 fp32 panel as fp16 into LDS (8 halfs / thread)
// ---------------------------------------------------------------------------
__device__ __forceinline__ void stage_a_f32(_Float16* at, const float* __restrict__ src,
                                            int ld, int tid) {
    const int idx = tid * 8;
    const int row = idx >> 7, col = idx & 127;
    const float* sp = &src[(size_t)row * ld + col];
    union { v8h v; _Float16 e[8]; } u;
    #pragma unroll
    for (int j = 0; j < 8; ++j) u.e[j] = (_Float16)sp[j];
    *(v8h*)&at[idx] = u.v;
}

// A fragment for K-chunk c from a 16x128 LDS panel (contiguous k order: k = 16*lhi + s)
__device__ __forceinline__ v16h afrag(const _Float16* at, int lm, int lhi, int c) {
    union { v16h v; v8h h[2]; } ua;
    ua.h[0] = *(const v8h*)&at[lm * 128 + c * 32 + lhi * 16];
    ua.h[1] = *(const v8h*)&at[lm * 128 + c * 32 + lhi * 16 + 8];
    return ua.v;
}

// ---------------------------------------------------------------------------
// 2) q = qr @ wq^T    (M=2048, N=4096, K=1536)
//    grid (128, 16), block 256 (8 waves). Block tile: 16 M x 256 N.
//    Each wave: 2 N-tiles, K unrolled by 128 -> 8 WMMAs per barrier pair.
// ---------------------------------------------------------------------------
__global__ void qi_qgemm(const float* __restrict__ qr, const _Float16* __restrict__ wq16,
                         float* __restrict__ qbuf) {
    const int mBase = blockIdx.x * 16;
    const int nBase = blockIdx.y * 256;
    const int tid  = threadIdx.x;
    const int wave = tid >> 5;
    const int lane = tid & 31;
    const int lhi  = (lane >> 4) & 1;
    const int lm   = lane & 15;
    __shared__ _Float16 at[16 * 128];
    v8f acc0 = {}, acc1 = {};
    const int o0 = nBase + wave * 16 + lm;
    const int o1 = o0 + 128;
    for (int k0 = 0; k0 < QR_; k0 += 128) {
        stage_a_f32(at, &qr[(size_t)mBase * QR_ + k0], QR_, tid);
        __syncthreads();
        #pragma unroll
        for (int c = 0; c < 4; ++c) {
            v16h a  = afrag(at, lm, lhi, c);
            v16h b0 = *(const v16h*)&wq16[(size_t)o0 * QR_ + k0 + c * 32 + lhi * 16];
            v16h b1 = *(const v16h*)&wq16[(size_t)o1 * QR_ + k0 + c * 32 + lhi * 16];
            acc0 = WMMA_F16(a, b0, acc0);
            acc1 = WMMA_F16(a, b1, acc1);
        }
        __syncthreads();
    }
    #pragma unroll
    for (int r = 0; r < 8; ++r) {
        const int m = mBase + r + lhi * 8;
        qbuf[(size_t)m * (NH * HD) + o0] = acc0[r];
        qbuf[(size_t)m * (NH * HD) + o1] = acc1[r];
    }
}

// ---------------------------------------------------------------------------
// 3) k = x @ wk^T    (M=2048, N=128, K=4096), K unrolled by 128.
// ---------------------------------------------------------------------------
__global__ void qi_kgemm(const float* __restrict__ x, const _Float16* __restrict__ wk16,
                         float* __restrict__ kbuf) {
    const int mBase = blockIdx.x * 16;
    const int tid  = threadIdx.x;
    const int wave = tid >> 5;
    const int lane = tid & 31;
    const int lhi  = (lane >> 4) & 1;
    const int lm   = lane & 15;
    __shared__ _Float16 at[16 * 128];
    v8f acc = {};
    const int o = wave * 16 + lm;             // 0..127
    for (int k0 = 0; k0 < DIM; k0 += 128) {
        stage_a_f32(at, &x[(size_t)mBase * DIM + k0], DIM, tid);
        __syncthreads();
        #pragma unroll
        for (int c = 0; c < 4; ++c) {
            v16h a = afrag(at, lm, lhi, c);
            v16h b = *(const v16h*)&wk16[(size_t)o * DIM + k0 + c * 32 + lhi * 16];
            acc = WMMA_F16(a, b, acc);
        }
        __syncthreads();
    }
    #pragma unroll
    for (int r = 0; r < 8; ++r)
        kbuf[(size_t)(mBase + r + lhi * 8) * HD + o] = acc[r];
}

// ---------------------------------------------------------------------------
// Shared row postprocess helpers: RoPE + FWHT(128) + 4-bit fake-quant (fp16)
// ---------------------------------------------------------------------------
__device__ __forceinline__ float qi_rope(float v, const float* buf, int d, int t) {
    if (d < RD) {
        float inv = __powf(10000.0f, -(float)(d & 31) * (1.0f / 32.0f));
        float ang = (float)t * inv;
        float sn, cs;
        __sincosf(ang, &sn, &cs);
        float rh = (d < 32) ? -buf[d + 32] : buf[d - 32];
        v = v * cs + rh * sn;
    }
    return v;
}

__device__ __forceinline__ void qi_fwht_quant(float* buf, float* red, int d, float x,
                                              _Float16* dst) {
    __syncthreads();
    buf[d] = x;
    for (int len = 1; len < HD; len <<= 1) {
        __syncthreads();
        float a = buf[d], b = buf[d ^ len];
        __syncthreads();
        buf[d] = (d & len) ? (b - a) : (a + b);
    }
    float hv = (float)(_Float16)(buf[d] * INV_SQRT_HD);
    __syncthreads();
    red[d] = fabsf(hv);
    __syncthreads();
    for (int off = 64; off > 0; off >>= 1) {
        if (d < off) red[d] = fmaxf(red[d], red[d + off]);
        __syncthreads();
    }
    float m = red[0];
    float s = (m == 0.0f) ? 1.0f : (float)(_Float16)(m / 7.0f);
    float r = fminf(fmaxf(rintf(hv / s), -7.0f), 7.0f);
    dst[d] = (_Float16)(r * s);
}

// ---------------------------------------------------------------------------
// 4) q postprocess: per (t,h) row of 128: RoPE(first 64, pos=t) + FWHT + quant
// ---------------------------------------------------------------------------
__global__ void qi_qpost(const float* __restrict__ qbuf, _Float16* __restrict__ qfq) {
    const int rowId = blockIdx.x;             // t*32 + h
    const int t = rowId >> 5;
    const int h = rowId & 31;
    const int d = threadIdx.x;                // 0..127
    __shared__ float buf[HD];
    __shared__ float red[HD];
    float v = qbuf[(size_t)t * (NH * HD) + h * HD + d];
    buf[d] = v;
    __syncthreads();
    float x = qi_rope(v, buf, d, t);
    qi_fwht_quant(buf, red, d, x, &qfq[(size_t)rowId * HD]);
}

// ---------------------------------------------------------------------------
// 5) k postprocess: layernorm + RoPE(pos=s) + FWHT + quant, per row s
// ---------------------------------------------------------------------------
__global__ void qi_kpost(const float* __restrict__ kbuf, const float* __restrict__ kn_w,
                         const float* __restrict__ kn_b, _Float16* __restrict__ kfq) {
    const int srow = blockIdx.x;              // 0..2047
    const int d = threadIdx.x;                // 0..127
    __shared__ float buf[HD];
    __shared__ float red[HD];
    float v = kbuf[(size_t)srow * HD + d];
    red[d] = v;
    __syncthreads();
    for (int off = 64; off > 0; off >>= 1) {
        if (d < off) red[d] += red[d + off];
        __syncthreads();
    }
    const float mu = red[0] * (1.0f / 128.0f);
    __syncthreads();
    float c0 = v - mu;
    red[d] = c0 * c0;
    __syncthreads();
    for (int off = 64; off > 0; off >>= 1) {
        if (d < off) red[d] += red[d + off];
        __syncthreads();
    }
    const float var = red[0] * (1.0f / 128.0f);
    float vn = c0 * rsqrtf(var + 1e-5f) * kn_w[d] + kn_b[d];
    __syncthreads();
    buf[d] = vn;
    __syncthreads();
    float x = qi_rope(vn, buf, d, srow);
    qi_fwht_quant(buf, red, d, x, &kfq[(size_t)srow * HD]);
}

// ---------------------------------------------------------------------------
// 6) w = fp16(x @ wproj^T) * fp16(1/sqrt(32)) * fp16(1/sqrt(128)), stored f32
// ---------------------------------------------------------------------------
__global__ void qi_wgemm(const float* __restrict__ x, const float* __restrict__ wproj,
                         float* __restrict__ wbuf) {
    const int gid = blockIdx.x * blockDim.x + threadIdx.x;  // 65536
    const int t = gid >> 5;
    const int h = gid & 31;
    const float* xr = &x[(size_t)t * DIM];
    const float* wr = &wproj[(size_t)h * DIM];
    float acc = 0.0f;
    for (int k = 0; k < DIM; ++k) acc = fmaf(xr[k], wr[k], acc);
    _Float16 hv = (_Float16)acc;
    hv = hv * (_Float16)0.17677669529663687f;   // NH^-0.5 in fp16
    hv = hv * (_Float16)0.08838834764831845f;   // HD^-0.5 in fp16
    wbuf[gid] = (float)hv;
}

// ---------------------------------------------------------------------------
// 7) Fused index score:
//    score[t,s] = sum_h w[t,h] * relu(q_fq[t,h,:] . k_fq[s,:]) + causal(t,s)
//    grid (128, 8), block 256 (8 waves). Block: 16 t x 256 s.
//    Each wave holds TWO resident k-tiles in VGPRs (8 x v16h) across all heads.
//    q tile staged per head via GLOBAL_LOAD_ASYNC_TO_LDS (ASYNCcnt path).
// ---------------------------------------------------------------------------
__global__ void qi_score(const _Float16* __restrict__ qfq, const _Float16* __restrict__ kfq,
                         const float* __restrict__ wbuf, float* __restrict__ score) {
    const int tBase = blockIdx.x * 16;
    const int sBase = blockIdx.y * 256;
    const int tid  = threadIdx.x;
    const int wave = tid >> 5;
    const int lane = tid & 31;
    const int lhi  = (lane >> 4) & 1;
    const int lm   = lane & 15;
    const int sT0 = sBase + wave * 16;
    const int sT1 = sT0 + 128;
    __shared__ _Float16 qt[16 * HD];          // one head's 16 x 128 q tile
    __shared__ float    wl[16 * NH];          // w[tBase..tBase+15][0..31]
    for (int i = tid; i < 16 * NH; i += 256)
        wl[i] = wbuf[(tBase + (i >> 5)) * NH + (i & 31)];
    // resident B fragments (k rows are K-contiguous)
    v16h bf0[4], bf1[4];
    {
        const int s0 = sT0 + lm, s1 = sT1 + lm;
        #pragma unroll
        for (int c = 0; c < 4; ++c) {
            bf0[c] = *(const v16h*)&kfq[(size_t)s0 * HD + c * 32 + lhi * 16];
            bf1[c] = *(const v16h*)&kfq[(size_t)s1 * HD + c * 32 + lhi * 16];
        }
    }
    // per-thread async staging addresses: 8 halfs (16B) each
    const int sidx = tid * 8;
    const int sm = sidx >> 7, sd = sidx & 127;
    const unsigned lds_off = (unsigned)(uintptr_t)&qt[sidx];
    float acc0[8] = {0, 0, 0, 0, 0, 0, 0, 0};
    float acc1[8] = {0, 0, 0, 0, 0, 0, 0, 0};
    for (int h = 0; h < NH; ++h) {
        __syncthreads();   // previous iteration's fragments consumed
        {   // async copy q tile for this head into LDS (16B per lane)
            const _Float16* gp = &qfq[((size_t)(tBase + sm) * NH + h) * HD + sd];
            asm volatile("global_load_async_to_lds_b128 %0, %1, off"
                         :: "v"(lds_off), "v"(gp) : "memory");
            asm volatile("s_wait_asynccnt 0" ::: "memory");
        }
        __syncthreads();
        v8f t0 = {}, t1 = {};
        #pragma unroll
        for (int c = 0; c < 4; ++c) {
            v16h a = afrag(qt, lm, lhi, c);
            t0 = WMMA_F16(a, bf0[c], t0);
            t1 = WMMA_F16(a, bf1[c], t1);
        }
        #pragma unroll
        for (int r = 0; r < 8; ++r) {
            float wv = wl[(r + lhi * 8) * NH + h];
            acc0[r] += fmaxf(t0[r], 0.0f) * wv;
            acc1[r] += fmaxf(t1[r], 0.0f) * wv;
        }
    }
    const int s0 = sT0 + lm, s1 = sT1 + lm;
    #pragma unroll
    for (int r = 0; r < 8; ++r) {
        const int tt = tBase + r + lhi * 8;
        score[(size_t)tt * S_ + s0] = acc0[r] + ((s0 <= tt) ? 0.0f : NEG_INF_MASK);
        score[(size_t)tt * S_ + s1] = acc1[r] + ((s1 <= tt) ? 0.0f : NEG_INF_MASK);
    }
}

// ---------------------------------------------------------------------------
// 8) Top-1024 of each 2048-row: LDS bitonic sort, descending, idx tie-break.
// ---------------------------------------------------------------------------
__global__ void qi_topk(const float* __restrict__ score, int* __restrict__ out) {
    const int t = blockIdx.x;
    __shared__ float sv[S_];
    __shared__ int   si[S_];
    for (int i = threadIdx.x; i < S_; i += 256) {
        sv[i] = score[(size_t)t * S_ + i];
        si[i] = i;
    }
    __syncthreads();
    for (int k = 2; k <= S_; k <<= 1) {
        for (int j = k >> 1; j > 0; j >>= 1) {
            for (int i = threadIdx.x; i < S_; i += 256) {
                const int ix = i ^ j;
                if (ix > i) {
                    const bool up = ((i & k) == 0);
                    float a = sv[i], b = sv[ix];
                    int ai = si[i], bi = si[ix];
                    bool before = (b > a) || (b == a && bi < ai);
                    if (before == up) {
                        sv[i] = b; sv[ix] = a; si[i] = bi; si[ix] = ai;
                    }
                }
            }
            __syncthreads();
        }
    }
    for (int i = threadIdx.x; i < 1024; i += 256)
        out[(size_t)t * 1024 + i] = si[i];
}

// ---------------------------------------------------------------------------
extern "C" void kernel_launch(void* const* d_in, const int* in_sizes, int n_in,
                              void* d_out, int out_size, void* d_ws, size_t ws_size,
                              hipStream_t stream) {
    const float* x       = (const float*)d_in[0];   // (1,2048,4096)
    const float* qr      = (const float*)d_in[1];   // (1,2048,1536)
    // d_in[2] mask, d_in[3] position_ids: regenerated inline (causal / arange)
    const float* wq_b_w  = (const float*)d_in[4];   // (4096,1536)
    const float* wk_w    = (const float*)d_in[5];   // (128,4096)
    const float* kn_w    = (const float*)d_in[6];   // (128,)
    const float* kn_b    = (const float*)d_in[7];   // (128,)
    const float* wproj_w = (const float*)d_in[8];   // (32,4096)
    int* out = (int*)d_out;                          // (1,2048,1024) int32

    char* ws = (char*)d_ws;
    size_t off = 0;
    auto alloc = [&](size_t bytes) -> void* {
        void* p = ws + off;
        off = (off + bytes + 255) & ~(size_t)255;
        return p;
    };
    _Float16* wq16 = (_Float16*)alloc((size_t)4096 * QR_ * 2);   // 12.6 MB
    _Float16* wk16 = (_Float16*)alloc((size_t)HD * DIM * 2);     //  1.0 MB
    float*    qbuf = (float*)   alloc((size_t)S_ * NH * HD * 4); // 33.6 MB
    _Float16* qfq  = (_Float16*)alloc((size_t)S_ * NH * HD * 2); // 16.8 MB
    float*    kbuf = (float*)   alloc((size_t)S_ * HD * 4);      //  1.0 MB
    _Float16* kfq  = (_Float16*)alloc((size_t)S_ * HD * 2);      //  0.5 MB
    float*    wbuf = (float*)   alloc((size_t)S_ * NH * 4);      //  0.3 MB
    float*    scor = (float*)   alloc((size_t)S_ * S_ * 4);      // 16.8 MB

    qi_wq_quant<<<4096, 256, 0, stream>>>(wq_b_w, wq16);
    qi_wk16<<<(HD * DIM) / 256, 256, 0, stream>>>(wk_w, wk16);
    qi_qgemm<<<dim3(S_ / 16, (NH * HD) / 256), 256, 0, stream>>>(qr, wq16, qbuf);
    qi_qpost<<<S_ * NH, HD, 0, stream>>>(qbuf, qfq);
    qi_kgemm<<<S_ / 16, 256, 0, stream>>>(x, wk16, kbuf);
    qi_kpost<<<S_, HD, 0, stream>>>(kbuf, kn_w, kn_b, kfq);
    qi_wgemm<<<(S_ * NH) / 256, 256, 0, stream>>>(x, wproj_w, wbuf);
    qi_score<<<dim3(S_ / 16, S_ / 256), 256, 0, stream>>>(qfq, kfq, wbuf, scor);
    qi_topk<<<S_, 256, 0, stream>>>(scor, out);
}
